// Model_37383395344915
// MI455X (gfx1250) — matrix operand order repaired and checked by reading, not compiled, not run on hardware
//
#include <hip/hip_runtime.h>
#include <math.h>

typedef __attribute__((ext_vector_type(16))) _Float16 v16h;
typedef __attribute__((ext_vector_type(8)))  float    v8f;

// ---- problem constants ----
constexpr int kB   = 256;
constexpr int kL   = 128;
constexpr int kE   = 300;   // embedding dim
constexpr int kH   = 75;    // hidden per direction
constexpr int k4H  = 300;   // 4*H gate width
constexpr int KE   = 320;   // E padded to mult of 32 (WMMA K)
constexpr int N4H  = 320;   // 4H padded to 20 n-tiles (uniform loops)
constexpr int KH   = 96;    // H padded to mult of 32
constexpr int RP   = 160;   // REPR(150) padded to mult of 32
constexpr int MROWS = kB * kL;  // 32768

// fragment geometry: one WMMA B tile = 32 lanes x 16 halves = 512 halves
constexpr int FR    = 512;
constexpr int NT4H  = N4H / 16;  // 20
constexpr int KS_E  = KE / 32;   // 10
constexpr int KS_H  = KH / 32;   // 3
constexpr int NT_U  = RP / 16;   // 10
constexpr int KS_U  = RP / 32;   // 5

// ---- workspace layout ----
constexpr size_t SZH_WIH = (size_t)2 * NT4H * KS_E * FR;   // 204800 halves
constexpr size_t SZH_WHH = (size_t)2 * NT4H * KS_H * FR;   //  61440
constexpr size_t SZH_U   = (size_t)NT_U * KS_U * FR;       //  25600
constexpr size_t SZH_R2F = (size_t)kB * 8 * KS_U * FR;     // 5242880
constexpr int    SZ_BIAS = 2 * N4H;                        // 640
constexpr size_t SZ_G    = (size_t)4 * MROWS * N4H;
constexpr size_t SZ_R    = (size_t)2 * MROWS * RP;
constexpr size_t SZ_S    = (size_t)MROWS * RP;
constexpr size_t SZ_GM   = (size_t)kB * kL * kL;

union FragU { uint4 q[2]; v16h v; };

__device__ __forceinline__ v8f vzero8() {
  v8f z = {0.f,0.f,0.f,0.f,0.f,0.f,0.f,0.f};
  return z;
}

__device__ __forceinline__ v8f wmma_f16(v16h a, v16h b, v8f c) {
  return __builtin_amdgcn_wmma_f32_16x16x32_f16(false, a, false, b,
                                                (short)0, c, false, false);
}

// load a pre-swizzled B fragment: lane's 16 halves are contiguous (2x b128)
__device__ __forceinline__ v16h load_frag(const _Float16* frag, int lane) {
  FragU f;
  const uint4* p = (const uint4*)(frag) + lane * 2;
  f.q[0] = p[0];
  f.q[1] = p[1];
  return f.v;
}

// A tile from an f32 row, scaled by vm (0/1 lane mask; branch-free):
// halves 0..7 <- row[kb..kb+7]*vm, halves 8..15 <- row[kb+16..kb+23]*vm
__device__ __forceinline__ v16h load_a_f32v(const float* row, int kb, float vm) {
  const float4 x0 = *(const float4*)(row + kb);
  const float4 x1 = *(const float4*)(row + kb + 4);
  const float4 y0 = *(const float4*)(row + kb + 16);
  const float4 y1 = *(const float4*)(row + kb + 20);
  v16h a;
  a[0]=(_Float16)(x0.x*vm); a[1]=(_Float16)(x0.y*vm);
  a[2]=(_Float16)(x0.z*vm); a[3]=(_Float16)(x0.w*vm);
  a[4]=(_Float16)(x1.x*vm); a[5]=(_Float16)(x1.y*vm);
  a[6]=(_Float16)(x1.z*vm); a[7]=(_Float16)(x1.w*vm);
  a[8]=(_Float16)(y0.x*vm); a[9]=(_Float16)(y0.y*vm);
  a[10]=(_Float16)(y0.z*vm); a[11]=(_Float16)(y0.w*vm);
  a[12]=(_Float16)(y1.x*vm); a[13]=(_Float16)(y1.y*vm);
  a[14]=(_Float16)(y1.z*vm); a[15]=(_Float16)(y1.w*vm);
  return a;
}

// A tile from an f16 LDS row (2x ds_load_b128)
__device__ __forceinline__ v16h load_a_f16_lds(const _Float16* row, int kb) {
  FragU f;
  f.q[0] = *(const uint4*)(row + kb);
  f.q[1] = *(const uint4*)(row + kb + 16);
  return f.v;
}

// ---------------------------------------------------------------------------
// K0: zero fill
// ---------------------------------------------------------------------------
__global__ void zero_f32(float* p, size_t n) {
  for (size_t i = (size_t)blockIdx.x * blockDim.x + threadIdx.x; i < n;
       i += (size_t)gridDim.x * blockDim.x)
    p[i] = 0.f;
}

// ---------------------------------------------------------------------------
// K1: pack weights into pre-swizzled f16 WMMA B fragments (+ f32 bias)
//     fragment element (lane,h): k = ks*32 + (lane>>4)*8 + (h&7) + (h>>3)*16,
//                                n = nt*16 + (lane&15)
// ---------------------------------------------------------------------------
__global__ void pack_weights(const float* Wih_f, const float* Whh_f,
                             const float* bih_f, const float* bhh_f,
                             const float* Wih_b, const float* Whh_b,
                             const float* bih_b, const float* bhh_b,
                             const float* u,
                             _Float16* wihF, _Float16* whhF, _Float16* uF,
                             float* bias) {
  const size_t totalH = SZH_WIH + SZH_WHH + SZH_U;
  const size_t total  = totalH + SZ_BIAS;
  for (size_t idx = (size_t)blockIdx.x * blockDim.x + threadIdx.x; idx < total;
       idx += (size_t)gridDim.x * blockDim.x) {
    if (idx >= totalH) {                       // bias (f32)
      const int i = (int)(idx - totalH);
      const int dir = i / N4H, n = i % N4H;
      const float* p1 = dir ? bih_b : bih_f;
      const float* p2 = dir ? bhh_b : bhh_f;
      bias[i] = (n < k4H) ? (p1[n] + p2[n]) : 0.f;
      continue;
    }
    size_t i = idx;
    const int h    = (int)(i & 15);
    const int lane = (int)((i >> 4) & 31);
    size_t frag    = i >> 9;
    const int kloc = ((lane >> 4) << 3) + (h & 7) + ((h >> 3) << 4);
    const int nloc = lane & 15;
    if (i < SZH_WIH) {
      const int ks = (int)(frag % KS_E); frag /= KS_E;
      const int nt = (int)(frag % NT4H);
      const int dir = (int)(frag / NT4H);
      const int k = ks * 32 + kloc, n = nt * 16 + nloc;
      const float* W = dir ? Wih_b : Wih_f;
      wihF[i] = (k < kE && n < k4H) ? (_Float16)W[n * kE + k] : (_Float16)0.f;
      continue;
    }
    i -= SZH_WIH;
    if (i < SZH_WHH) {
      frag = i >> 9;
      const int ks = (int)(frag % KS_H); frag /= KS_H;
      const int nt = (int)(frag % NT4H);
      const int dir = (int)(frag / NT4H);
      const int k = ks * 32 + kloc, n = nt * 16 + nloc;
      const float* W = dir ? Whh_b : Whh_f;
      whhF[i] = (k < kH && n < k4H) ? (_Float16)W[n * kH + k] : (_Float16)0.f;
      continue;
    }
    i -= SZH_WHH;
    {
      frag = i >> 9;
      const int ks = (int)(frag % KS_U);
      const int nt = (int)(frag / KS_U);
      const int k = ks * 32 + kloc, n = nt * 16 + nloc;
      uF[i] = (k < 2 * kH && n < 2 * kH) ? (_Float16)u[k * (2 * kH) + n]
                                         : (_Float16)0.f;
    }
  }
}

// ---------------------------------------------------------------------------
// K2: input projection GEMM with fused embedding gather (branch-free body)
//     grid = (5 n-groups, 256 m-blocks, 4 combos), block = 256 (8 waves)
// ---------------------------------------------------------------------------
__global__ void input_proj(const int* q1, const int* q2,
                           const int* len1, const int* len2,
                           const float* emb, const _Float16* wihF, float* G) {
  const int lane = threadIdx.x & 31;
  const int wave = threadIdx.x >> 5;
  const int combo = blockIdx.z;
  const int qsel = combo >> 1, dir = combo & 1;
  const int* tok = qsel ? q2 : q1;
  const int* len = qsel ? len2 : len1;

  const int rowBase = blockIdx.y * 128 + wave * 16;
  const int row = rowBase + (lane & 15);   // A-tile row of this lane
  const int b = row >> 7, t = row & 127;
  int tcol = t;
  bool valid = true;
  if (dir) { tcol = len[b] - 1 - t; valid = (tcol >= 0); }
  const float vm = valid ? 1.f : 0.f;
  const int tc = valid ? tcol : 0;
  const float* erow = emb + (size_t)tok[b * kL + tc] * kE;

  const _Float16* WdF =
      wihF + ((size_t)dir * NT4H + blockIdx.x * 4) * KS_E * FR;
  const int nt0 = blockIdx.x * 4;          // 4 N-tiles per wave (20 total)

  v8f acc[4];
#pragma unroll
  for (int j = 0; j < 4; ++j) acc[j] = vzero8();

  for (int ks = 0; ks < KS_E; ++ks) {
    // ---- A tile ----
    const int kb = ks * 32 + ((lane >> 4) << 3);
    v16h a;
    if (ks < KS_E - 1) {                 // fully in-range: vector loads
      a = load_a_f32v(erow, kb, vm);
    } else {                             // tail k-step straddles E=300
#pragma unroll
      for (int i = 0; i < 8; ++i) {
        const int k0 = kb + i;
        a[i]     = (_Float16)((k0 < kE ? erow[k0] : 0.f) * vm);
        a[i + 8] = (_Float16)0.f;        // kb+16+i >= 304 > 299 always here
      }
    }
    // ---- 4 B fragments + WMMA (no guards: NT4H == 20) ----
#pragma unroll
    for (int j = 0; j < 4; ++j) {
      const v16h bv = load_frag(WdF + ((size_t)j * KS_E + ks) * FR, lane);
      acc[j] = wmma_f16(a, bv, acc[j]);
    }
  }

  float* Gc = G + (size_t)combo * MROWS * N4H;
  const int mo = (lane >> 4) << 3;
  const int c0 = lane & 15;
#pragma unroll
  for (int j = 0; j < 4; ++j)
#pragma unroll
    for (int r = 0; r < 8; ++r)
      Gc[(size_t)(rowBase + mo + r) * N4H + (nt0 + j) * 16 + c0] = acc[j][r];
}

// ---------------------------------------------------------------------------
// K3: persistent LSTM scan. One block per (combo, 32-batch tile).
//     Whh lives in LDS as pre-swizzled f16 fragments; h state in LDS (f16);
//     c state in registers. Uniform 5-job loop per wave (2*20 tiles / 8).
// ---------------------------------------------------------------------------
__global__ void lstm_scan(const int* len1, const int* len2,
                          const float* G, const _Float16* whhF,
                          const float* bias, float* R) {
  extern __shared__ char smemRaw[];
  _Float16* sWhh  = (_Float16*)smemRaw;                 // NT4H*KS_H*FR f16
  _Float16* sH    = sWhh + NT4H * KS_H * FR;            // 32*KH  f16
  float*    sGate = (float*)(sH + 32 * KH);             // 32*N4H f32
  int*      sLen  = (int*)(sGate + 32 * N4H);           // 32

  const int tid = threadIdx.x;
  const int lane = tid & 31, wave = tid >> 5;
  const int combo = blockIdx.y, qsel = combo >> 1, dir = combo & 1;
  const int btile = blockIdx.x;  // 32 batch rows per block

  const float* Gc = G + (size_t)combo * MROWS * N4H;
  const float* bi = bias + dir * N4H;
  float* Rq = R + (size_t)qsel * MROWS * RP;
  const int* len = qsel ? len2 : len1;

  // straight vector copy of pre-swizzled Whh fragments into LDS
  {
    const uint4* src = (const uint4*)(whhF + (size_t)dir * NT4H * KS_H * FR);
    uint4* dst = (uint4*)sWhh;
    for (int i = tid; i < NT4H * KS_H * FR / 8; i += blockDim.x) dst[i] = src[i];
  }
  for (int i = tid; i < 32 * KH; i += blockDim.x) sH[i] = (_Float16)0.f;
  if (tid < 32) sLen[tid] = len[btile * 32 + tid];
  __syncthreads();

  float cReg[10];
#pragma unroll
  for (int j = 0; j < 10; ++j) cReg[j] = 0.f;

  for (int t = 0; t < kL; ++t) {
    // ---- recurrent GEMM: 2 row-tiles x 20 n-tiles over 8 waves, uniform ----
#pragma unroll
    for (int jj = 0; jj < 2 * NT4H / 8; ++jj) {
      const int job = wave + jj * 8;
      const int mt = job / NT4H;
      const int nt = job % NT4H;
      v8f acc = vzero8();
      const int arow = mt * 16 + (lane & 15);
#pragma unroll
      for (int ks = 0; ks < KS_H; ++ks) {
        const int kb = ks * 32 + ((lane >> 4) << 3);
        const v16h a  = load_a_f16_lds(sH + arow * KH, kb);
        const v16h bv = load_frag(sWhh + (nt * KS_H + ks) * FR, lane);
        acc = wmma_f16(a, bv, acc);
      }
      const int col = nt * 16 + (lane & 15);
      const int mo = (lane >> 4) << 3;
#pragma unroll
      for (int r = 0; r < 8; ++r) {
        const int lrow = mt * 16 + mo + r;
        const size_t grow = (size_t)(btile * 32 + lrow) * kL + t;
        sGate[lrow * N4H + col] = acc[r] + Gc[grow * N4H + col] + bi[col];
      }
      // prefetch next timestep's gate row for this tile (global_prefetch_b8)
      if (t + 1 < kL) {
        const size_t gnext = (size_t)(btile * 32 + mt * 16 + mo) * kL + (t + 1);
        __builtin_prefetch(&Gc[gnext * N4H + col], 0, 1);
      }
    }
    __syncthreads();

    // ---- elementwise cell (torch gate order i,f,g,o) ----
    int j = 0;
    for (int e = tid; e < 32 * kH; e += blockDim.x, ++j) {
      const int rw = e / kH, cl = e % kH;
      const float gi = sGate[rw * N4H + cl];
      const float gf = sGate[rw * N4H + kH + cl];
      const float gg = sGate[rw * N4H + 2 * kH + cl];
      const float go = sGate[rw * N4H + 3 * kH + cl];
      const float si = 1.f / (1.f + __expf(-gi));
      const float sf = 1.f / (1.f + __expf(-gf));
      const float so = 1.f / (1.f + __expf(-go));
      const float c = sf * cReg[j] + si * tanhf(gg);
      cReg[j] = c;
      const float h = so * tanhf(c);
      sH[rw * KH + cl] = (_Float16)h;
      const int bg = btile * 32 + rw;
      const int ln = sLen[rw];
      if (!dir) {
        if (t < ln) Rq[((size_t)bg * kL + t) * RP + cl] = h;
      } else {
        const int pos = ln - 1 - t;
        if (pos >= 0) Rq[((size_t)bg * kL + pos) * RP + kH + cl] = h;
      }
    }
    __syncthreads();
  }
}

// ---------------------------------------------------------------------------
// K4: S = r1 @ u  (M=32768, K=160, N=160); A vectorized f32, B fragments
// ---------------------------------------------------------------------------
__global__ void sgemm_ru(const float* R, const _Float16* uF, float* S) {
  const int lane = threadIdx.x & 31, wave = threadIdx.x >> 5;
  const int rowBase = blockIdx.y * 128 + wave * 16;
  const int nt0 = blockIdx.x * 5;   // 10 n-tiles total
  v8f acc[5];
#pragma unroll
  for (int j = 0; j < 5; ++j) acc[j] = vzero8();
  const float* arow = R + (size_t)(rowBase + (lane & 15)) * RP;
  for (int ks = 0; ks < KS_U; ++ks) {
    const int kb = ks * 32 + ((lane >> 4) << 3);
    const v16h a = load_a_f32v(arow, kb, 1.f);
#pragma unroll
    for (int j = 0; j < 5; ++j) {
      const v16h bv =
          load_frag(uF + ((size_t)(nt0 + j) * KS_U + ks) * FR, lane);
      acc[j] = wmma_f16(a, bv, acc[j]);
    }
  }
  const int mo = (lane >> 4) << 3, c0 = lane & 15;
#pragma unroll
  for (int j = 0; j < 5; ++j)
#pragma unroll
    for (int r = 0; r < 8; ++r)
      S[(size_t)(rowBase + mo + r) * RP + (nt0 + j) * 16 + c0] = acc[j][r];
}

// ---------------------------------------------------------------------------
// K4b: swizzle r2 into f16 B fragments for the attention GEMM
//      r2F[b][nt][ks][lane][h] = r2[b][nt*16+(lane&15)][ks*32+(lane>>4)*8+(h&7)+(h>>3)*16]
// ---------------------------------------------------------------------------
__global__ void swizzle_r2(const float* R, _Float16* r2F) {
  const float* R2 = R + (size_t)MROWS * RP;
  for (size_t idx = (size_t)blockIdx.x * blockDim.x + threadIdx.x;
       idx < SZH_R2F; idx += (size_t)gridDim.x * blockDim.x) {
    const int h    = (int)(idx & 15);
    const int lane = (int)((idx >> 4) & 31);
    size_t frag    = idx >> 9;
    const int ks = (int)(frag % KS_U); frag /= KS_U;
    const int nt = (int)(frag % 8);
    const int b  = (int)(frag / 8);
    const int n = nt * 16 + (lane & 15);
    const int k = ks * 32 + ((lane >> 4) << 3) + (h & 7) + ((h >> 3) << 4);
    r2F[idx] = (_Float16)R2[((size_t)b * kL + n) * RP + k];
  }
}

// ---------------------------------------------------------------------------
// K5: per-batch g = tanh(S @ r2^T) with pair mask folded in (-1e30 outside)
// ---------------------------------------------------------------------------
__global__ void gmat(const float* S, const _Float16* r2F,
                     const int* len1, const int* len2, float* GM) {
  const int b = blockIdx.x;
  const int lane = threadIdx.x & 31, wave = threadIdx.x >> 5;
  const float* Sb = S + (size_t)b * kL * RP;
  const int L1 = len1[b], L2 = len2[b];
  const int rowT = wave * 16;
  v8f acc[8];
#pragma unroll
  for (int j = 0; j < 8; ++j) acc[j] = vzero8();
  const float* arow = Sb + (size_t)(rowT + (lane & 15)) * RP;
  for (int ks = 0; ks < KS_U; ++ks) {
    const int kb = ks * 32 + ((lane >> 4) << 3);
    const v16h a = load_a_f32v(arow, kb, 1.f);
#pragma unroll
    for (int j = 0; j < 8; ++j) {
      const v16h bv =
          load_frag(r2F + (((size_t)b * 8 + j) * KS_U + ks) * FR, lane);
      acc[j] = wmma_f16(a, bv, acc[j]);
    }
  }
  float* Gb = GM + (size_t)b * kL * kL;
  const int mo = (lane >> 4) << 3, c0 = lane & 15;
#pragma unroll
  for (int j = 0; j < 8; ++j)
#pragma unroll
    for (int r = 0; r < 8; ++r) {
      const int l = rowT + mo + r, m = j * 16 + c0;
      Gb[l * kL + m] = (l < L1 && m < L2) ? tanhf(acc[j][r]) : -1e30f;
    }
}

// ---------------------------------------------------------------------------
// K6: attention finish + MLP head. One block per batch row.
// ---------------------------------------------------------------------------
__global__ void attn_finish(const float* GM, const float* R,
                            const int* len1, const int* len2,
                            const float* W1, const float* b1,
                            const float* W2, const float* b2, float* out) {
  const int b = blockIdx.x, tid = threadIdx.x;
  __shared__ float aw1[kL], aw2[kL];
  __shared__ float joint[2 * 2 * kH];   // 300
  __shared__ float t1s[300];
  const float* Gb = GM + (size_t)b * kL * kL;
  const int L1 = len1[b], L2 = len2[b];

  if (tid < kL) {
    float mr = -1e30f, mc = -1e30f;
    for (int i = 0; i < kL; ++i) {
      mr = fmaxf(mr, Gb[tid * kL + i]);
      mc = fmaxf(mc, Gb[i * kL + tid]);
    }
    aw1[tid] = (tid < L1) ? mr : -1e30f;
    aw2[tid] = (tid < L2) ? mc : -1e30f;
  }
  __syncthreads();
  if (tid == 0) {
    float mx = -1e30f;
    for (int i = 0; i < kL; ++i) mx = fmaxf(mx, aw1[i]);
    float s = 0.f;
    for (int i = 0; i < kL; ++i) { float e = expf(aw1[i] - mx); aw1[i] = e; s += e; }
    const float inv = 1.f / s;
    for (int i = 0; i < kL; ++i) aw1[i] *= inv;
  } else if (tid == 32) {
    float mx = -1e30f;
    for (int i = 0; i < kL; ++i) mx = fmaxf(mx, aw2[i]);
    float s = 0.f;
    for (int i = 0; i < kL; ++i) { float e = expf(aw2[i] - mx); aw2[i] = e; s += e; }
    const float inv = 1.f / s;
    for (int i = 0; i < kL; ++i) aw2[i] *= inv;
  }
  __syncthreads();
  if (tid < 2 * kH) {
    const float* R1b = R + (size_t)b * kL * RP;
    const float* R2b = R + (size_t)MROWS * RP + (size_t)b * kL * RP;
    float h1 = 0.f, h2 = 0.f;
    for (int l = 0; l < kL; ++l) {
      h1 += aw1[l] * R1b[(size_t)l * RP + tid];
      h2 += aw2[l] * R2b[(size_t)l * RP + tid];
    }
    joint[tid] = fabsf(h1 - h2);
    joint[2 * kH + tid] = h1 * h2;
  }
  __syncthreads();
  for (int j = tid; j < 300; j += blockDim.x) {
    float acc = b1[j];
    for (int i = 0; i < 300; ++i) acc += joint[i] * W1[j * 300 + i];
    t1s[j] = tanhf(acc);
  }
  __syncthreads();
  if (tid < 2) {
    float acc = b2[tid];
    for (int j = 0; j < 300; ++j) acc += t1s[j] * W2[tid * 300 + j];
    out[b * 2 + tid] = acc;
  }
}

// ---------------------------------------------------------------------------
extern "C" void kernel_launch(void* const* d_in, const int* in_sizes, int n_in,
                              void* d_out, int out_size, void* d_ws, size_t ws_size,
                              hipStream_t stream) {
  (void)in_sizes; (void)n_in; (void)out_size; (void)ws_size;
  const int*   q1    = (const int*)d_in[0];
  const int*   q2    = (const int*)d_in[1];
  const int*   len1  = (const int*)d_in[2];
  const int*   len2  = (const int*)d_in[3];
  const float* emb   = (const float*)d_in[4];
  const float* Wih_f = (const float*)d_in[5];
  const float* Whh_f = (const float*)d_in[6];
  const float* bih_f = (const float*)d_in[7];
  const float* bhh_f = (const float*)d_in[8];
  const float* Wih_b = (const float*)d_in[9];
  const float* Whh_b = (const float*)d_in[10];
  const float* bih_b = (const float*)d_in[11];
  const float* bhh_b = (const float*)d_in[12];
  const float* u     = (const float*)d_in[13];
  const float* W1    = (const float*)d_in[14];
  const float* b1    = (const float*)d_in[15];
  const float* W2    = (const float*)d_in[16];
  const float* b2    = (const float*)d_in[17];

  _Float16* wihF = (_Float16*)d_ws;
  _Float16* whhF = wihF + SZH_WIH;
  _Float16* uF   = whhF + SZH_WHH;
  _Float16* r2F  = uF + SZH_U;
  float* bias = (float*)(r2F + SZH_R2F);
  float* G    = bias + SZ_BIAS;
  float* R    = G + SZ_G;
  float* S    = R + SZ_R;
  float* GM   = S + SZ_S;

  // R must start zeroed (masked/padded positions are never written)
  zero_f32<<<2048, 256, 0, stream>>>(R, SZ_R);
  pack_weights<<<512, 256, 0, stream>>>(Wih_f, Whh_f, bih_f, bhh_f,
                                        Wih_b, Whh_b, bih_b, bhh_b, u,
                                        wihF, whhF, uF, bias);

  dim3 gProj(5, 256, 4);                 // n-groups, m-blocks, combos
  input_proj<<<gProj, 256, 0, stream>>>(q1, q2, len1, len2, emb, wihF, G);

  dim3 gScan(8, 4);                      // batch-tiles, combos
  const size_t smem = (size_t)NT4H * KS_H * FR * 2 + 32 * KH * 2 +
                      32 * N4H * 4 + 32 * 4;
  lstm_scan<<<gScan, 256, smem, stream>>>(len1, len2, G, whhF, bias, R);

  dim3 gS(2, 256);                       // n-groups, m-blocks
  sgemm_ru<<<gS, 256, 0, stream>>>(R, uF, S);

  swizzle_r2<<<2048, 256, 0, stream>>>(R, r2F);
  gmat<<<256, 256, 0, stream>>>(S, r2F, len1, len2, GM);

  attn_finish<<<256, 256, 0, stream>>>(GM, R, len1, len2, W1, b1, W2, b2,
                                       (float*)d_out);
}
